// SwitchSparseMoe_48455821033860
// MI455X (gfx1250) — compile-verified, matrix operand order checked by reading, not run on hardware
//
#include <hip/hip_runtime.h>
#include <hip/hip_bf16.h>
#include <stdint.h>

// ---------------- problem constants (match reference) ----------------
#define B 4
#define S 2048
#define DD 1024          // hidden dim D
#define FF 4096          // ffn dim F
#define E 8              // experts
#define CAP 320          // capacity per (batch, expert)

// ---------------- WMMA types ----------------
typedef __attribute__((ext_vector_type(16))) __bf16 v16bf;
typedef __attribute__((ext_vector_type(8)))  __bf16 v8bf;
typedef __attribute__((ext_vector_type(8)))  float  v8f;

// 128-bit payload type for the async global->LDS builtin (per hipcc diagnostic)
typedef int v4i __attribute__((vector_size(16)));
typedef __attribute__((address_space(1))) v4i* as1_v4i_p;
typedef __attribute__((address_space(3))) v4i* as3_v4i_p;

// ---------------- GEMM tiling ----------------
#define BM 128           // block rows   (3 M-tiles cover CAP=320)
#define BN 128           // block cols
#define BK 64            // K-slab per barrier
#define LDK (BK + 8)     // padded LDS row (bank-conflict-free fragment loads)
#define MT ((CAP + BM - 1) / BM)   // 3

// CDNA5 async global->LDS path (guarded: falls back to register staging)
#if __has_builtin(__builtin_amdgcn_global_load_async_to_lds_b128) && \
    __has_builtin(__builtin_amdgcn_s_wait_asynccnt)
#define USE_ASYNC_LDS 1
#else
#define USE_ASYNC_LDS 0
#endif

// ============================================================================
// Kernel 1: router — logits, softmax max-prob, argmax. One wave per token.
// ============================================================================
__global__ __launch_bounds__(256) void moe_router(
    const float* __restrict__ hidden, const float* __restrict__ gate_w,
    float* __restrict__ logits_out, int* __restrict__ expert_idx,
    float* __restrict__ top_prob)
{
    __shared__ float g[E * DD];                       // 32 KB
    for (int i = threadIdx.x; i < E * DD; i += 256) g[i] = gate_w[i];
    __syncthreads();

    const int wave  = threadIdx.x >> 5;
    const int lane  = threadIdx.x & 31;
    const int token = blockIdx.x * 8 + wave;          // grid = B*S/8

    const float* hp = hidden + (size_t)token * DD;
    float acc[E] = {};
    for (int d = lane; d < DD; d += 32) {
        float hv = hp[d];
#pragma unroll
        for (int e = 0; e < E; ++e) acc[e] += hv * g[e * DD + d];
    }
#pragma unroll
    for (int e = 0; e < E; ++e)
#pragma unroll
        for (int off = 16; off > 0; off >>= 1)
            acc[e] += __shfl_xor(acc[e], off, 32);

    if (lane == 0) {
        float m = acc[0]; int best = 0;
#pragma unroll
        for (int e = 1; e < E; ++e) if (acc[e] > m) { m = acc[e]; best = e; }
        float z = 0.f;
#pragma unroll
        for (int e = 0; e < E; ++e) z += __expf(acc[e] - m);
#pragma unroll
        for (int e = 0; e < E; ++e) logits_out[(size_t)token * E + e] = acc[e];
        expert_idx[token] = best;
        top_prob[token]   = 1.0f / z;                 // softmax prob of argmax
    }
}

// ============================================================================
// Kernel 2: capacity scan (sequential cumsum per (b,e); 32 threads, one wave)
// ============================================================================
__global__ void moe_route_scan(
    const int* __restrict__ expert_idx, int* __restrict__ toklist,
    int* __restrict__ count, int* __restrict__ eidx_out)
{
    int t = threadIdx.x;
    if (t >= B * E) return;
    int b = t >> 3, e = t & 7;
    int cnt = 0;
    for (int s = 0; s < S; ++s) {
        if (expert_idx[b * S + s] == e) {
            if (cnt < CAP) {
                toklist[(b * E + e) * CAP + cnt] = s;
                eidx_out[b * S + s] = e;
                ++cnt;
            } else {
                eidx_out[b * S + s] = 0;              // dropped -> expert 0
            }
        }
    }
    count[b * E + e] = cnt;
}

// ============================================================================
// Kernel 3: default output  out = top_prob * hidden  (dropped-token path;
// assigned tokens overwritten by GEMM2's scatter epilogue, same stream).
// ============================================================================
__global__ __launch_bounds__(256) void moe_default_out(
    const float* __restrict__ hidden, const float* __restrict__ top_prob,
    float* __restrict__ out)
{
    const int token = blockIdx.x;
    const float p = top_prob[token];
    const float* hp = hidden + (size_t)token * DD;
    float* op = out + (size_t)token * DD;
    for (int d = threadIdx.x; d < DD; d += 256) op[d] = p * hp[d];
}

// ============================================================================
// Kernel 4: GEMM1  h = silu(gather(hidden) @ w1[e]),  M=CAP N=FF K=DD
//   128x128 block tile, 32x64 per wave (8 WMMAs per 12 ds_loads),
//   register-prefetch + double-buffered LDS, one barrier per K-slab.
// ============================================================================
__global__ __launch_bounds__(256) void moe_gemm1(
    const float* __restrict__ hidden, const float* __restrict__ w1,
    const int* __restrict__ toklist, const int* __restrict__ count,
    __bf16* __restrict__ hbuf)
{
    const int group = blockIdx.z;                     // b*E + e
    const int b = group >> 3, e = group & 7;
    const int tileM = blockIdx.y, tileN = blockIdx.x;
    const int tid  = threadIdx.x;
    const int lane = tid & 31;
    const int wave = tid >> 5;

    __shared__ __bf16 As[2][BM][LDK];                 // [m][k]
    __shared__ __bf16 Bs[2][BN][LDK];                 // [n][k] fragment order

    const int cnt = count[group];
    const float* w1e = w1 + (size_t)e * DD * FF;

    v8f acc[2][4] = {};                               // 2 M-subtiles x 4 N-tiles
    const int wm = (wave & 3) * 32;                   // 4 wave rows of 32
    const int wn = (wave >> 2) * 64;                  // 2 wave col groups of 64

    // A: (row, 32-wide k segment) per thread: 128 rows x 2 segments
    const int ar = tid >> 1;                          // 0..127
    const int ak = (tid & 1) * 32;
    const float* asrc = nullptr;
    {
        int c = tileM * BM + ar;
        if (c < cnt)
            asrc = hidden + ((size_t)(b * S + toklist[group * CAP + c]) * DD + ak);
    }
    // B: 4 chunks of (8k x 1n) per thread
    int bn_[4], bk_[4];
#pragma unroll
    for (int i = 0; i < 4; ++i) {
        int li = tid + i * 256;
        bn_[i] = li & 127;
        bk_[i] = (li >> 7) * 8;
    }
    const float* bbase = w1e + tileN * BN;

    v8bf ra[4], rb[4];

    auto loadStage = [&](int k0) {
#pragma unroll
        for (int j = 0; j < 4; ++j) ra[j] = (v8bf){};
        if (asrc) {
            const float* s = asrc + k0;
#pragma unroll
            for (int j = 0; j < 4; ++j)
#pragma unroll
                for (int i = 0; i < 8; ++i) ra[j][i] = (__bf16)s[j * 8 + i];
        }
#pragma unroll
        for (int i = 0; i < 4; ++i) {
            const float* s = bbase + (size_t)(k0 + bk_[i]) * FF + bn_[i];
#pragma unroll
            for (int k = 0; k < 8; ++k) rb[i][k] = (__bf16)s[(size_t)k * FF];
        }
    };
    auto storeStage = [&](int bi) {
#pragma unroll
        for (int j = 0; j < 4; ++j) *(v8bf*)&As[bi][ar][ak + 8 * j] = ra[j];
#pragma unroll
        for (int i = 0; i < 4; ++i) *(v8bf*)&Bs[bi][bn_[i]][bk_[i]] = rb[i];
    };
    auto compute = [&](int bi) {
#pragma unroll
        for (int u = 0; u < 2; ++u) {
            const int kb = u * 32 + (lane >> 4) * 8;
            v16bf a[2];
#pragma unroll
            for (int mi = 0; mi < 2; ++mi) {
                int m = wm + mi * 16 + (lane & 15);
#pragma unroll
                for (int i = 0; i < 8; ++i) a[mi][i]     = As[bi][m][kb + i];
#pragma unroll
                for (int i = 0; i < 8; ++i) a[mi][8 + i] = As[bi][m][kb + 16 + i];
            }
#pragma unroll
            for (int t = 0; t < 4; ++t) {
                v16bf bf;
                int n = wn + t * 16 + (lane & 15);
#pragma unroll
                for (int i = 0; i < 8; ++i) bf[i]     = Bs[bi][n][kb + i];
#pragma unroll
                for (int i = 0; i < 8; ++i) bf[8 + i] = Bs[bi][n][kb + 16 + i];
#pragma unroll
                for (int mi = 0; mi < 2; ++mi)
                    acc[mi][t] = __builtin_amdgcn_wmma_f32_16x16x32_bf16(
                        false, a[mi], false, bf, (short)0, acc[mi][t],
                        false, false);
            }
        }
    };

    constexpr int NIT = DD / BK;                      // 16
    loadStage(0);
    storeStage(0);
    __syncthreads();
#pragma unroll 2
    for (int it = 0; it < NIT; ++it) {
        if (it + 1 < NIT) loadStage((it + 1) * BK);   // overlaps compute below
        compute(it & 1);
        if (it + 1 < NIT) {
            storeStage((it + 1) & 1);
            __syncthreads();
        }
    }

    // ---- epilogue: silu -> bf16 -> hbuf[group][c][f] (guard m < CAP) ----
#pragma unroll
    for (int mi = 0; mi < 2; ++mi) {
        const int mb = tileM * BM + wm + mi * 16 + (lane >> 4) * 8;
#pragma unroll
        for (int t = 0; t < 4; ++t) {
            int n = tileN * BN + wn + t * 16 + (lane & 15);
#pragma unroll
            for (int r = 0; r < 8; ++r) {
                int m = mb + r;
                if (m < CAP) {
                    float x = acc[mi][t][r];
                    float sv = x / (1.0f + __expf(-x));   // silu
                    hbuf[((size_t)group * CAP + m) * FF + n] = (__bf16)sv;
                }
            }
        }
    }
}

// ============================================================================
// Kernel 5: GEMM2  out[token] = top_prob * (h @ w2[e]),  M=CAP N=DD K=FF
//   A operand is bf16 in global memory: async global->LDS (ASYNCcnt path).
// ============================================================================
__global__ __launch_bounds__(256) void moe_gemm2(
    const __bf16* __restrict__ hbuf, const float* __restrict__ w2,
    const int* __restrict__ toklist, const int* __restrict__ count,
    const float* __restrict__ top_prob, float* __restrict__ out)
{
    const int group = blockIdx.z;
    const int b = group >> 3, e = group & 7;
    const int tileM = blockIdx.y, tileN = blockIdx.x;
    const int tid  = threadIdx.x;
    const int lane = tid & 31;
    const int wave = tid >> 5;

    __shared__ __bf16 As[2][BM][LDK];
    __shared__ __bf16 Bs[2][BN][LDK];

    const int cnt = count[group];
    const float* w2e = w2 + (size_t)e * FF * DD;

    v8f acc[2][4] = {};
    const int wm = (wave & 3) * 32;
    const int wn = (wave >> 2) * 64;

    const int ar = tid >> 1;
    const int ak = (tid & 1) * 32;
    const int ac = tileM * BM + ar;
    const __bf16* asrc =
        (ac < cnt) ? hbuf + ((size_t)group * CAP + ac) * FF + ak : nullptr;

    int bn_[4], bk_[4];
#pragma unroll
    for (int i = 0; i < 4; ++i) {
        int li = tid + i * 256;
        bn_[i] = li & 127;
        bk_[i] = (li >> 7) * 8;
    }
    const float* bbase = w2e + tileN * BN;

    v8bf ra[4], rb[4];

    auto loadStage = [&](int k0) {
#if !USE_ASYNC_LDS
#pragma unroll
        for (int j = 0; j < 4; ++j) ra[j] = (v8bf){};
        if (asrc) {
#pragma unroll
            for (int j = 0; j < 4; ++j) ra[j] = *(const v8bf*)(asrc + k0 + 8 * j);
        }
#endif
#pragma unroll
        for (int i = 0; i < 4; ++i) {
            const float* s = bbase + (size_t)(k0 + bk_[i]) * DD + bn_[i];
#pragma unroll
            for (int k = 0; k < 8; ++k) rb[i][k] = (__bf16)s[(size_t)k * DD];
        }
    };
    auto storeStage = [&](int bi, int k0) {
#if USE_ASYNC_LDS
        if (asrc) {
            // four async b128 copies: global bf16 tile -> LDS (ASYNCcnt)
#pragma unroll
            for (int j = 0; j < 4; ++j)
                __builtin_amdgcn_global_load_async_to_lds_b128(
                    (as1_v4i_p)(asrc + k0 + 8 * j),
                    (as3_v4i_p)&As[bi][ar][ak + 8 * j], 0, 0);
        } else {
            v8bf z = (v8bf){};
#pragma unroll
            for (int j = 0; j < 4; ++j) *(v8bf*)&As[bi][ar][ak + 8 * j] = z;
        }
#else
        (void)k0;
#pragma unroll
        for (int j = 0; j < 4; ++j) *(v8bf*)&As[bi][ar][ak + 8 * j] = ra[j];
#endif
#pragma unroll
        for (int i = 0; i < 4; ++i) *(v8bf*)&Bs[bi][bn_[i]][bk_[i]] = rb[i];
    };
    auto compute = [&](int bi) {
#pragma unroll
        for (int u = 0; u < 2; ++u) {
            const int kb = u * 32 + (lane >> 4) * 8;
            v16bf a[2];
#pragma unroll
            for (int mi = 0; mi < 2; ++mi) {
                int m = wm + mi * 16 + (lane & 15);
#pragma unroll
                for (int i = 0; i < 8; ++i) a[mi][i]     = As[bi][m][kb + i];
#pragma unroll
                for (int i = 0; i < 8; ++i) a[mi][8 + i] = As[bi][m][kb + 16 + i];
            }
#pragma unroll
            for (int t = 0; t < 4; ++t) {
                v16bf bf;
                int n = wn + t * 16 + (lane & 15);
#pragma unroll
                for (int i = 0; i < 8; ++i) bf[i]     = Bs[bi][n][kb + i];
#pragma unroll
                for (int i = 0; i < 8; ++i) bf[8 + i] = Bs[bi][n][kb + 16 + i];
#pragma unroll
                for (int mi = 0; mi < 2; ++mi)
                    acc[mi][t] = __builtin_amdgcn_wmma_f32_16x16x32_bf16(
                        false, a[mi], false, bf, (short)0, acc[mi][t],
                        false, false);
            }
        }
    };

    constexpr int NIT = FF / BK;                      // 64
    loadStage(0);
    storeStage(0, 0);
#if USE_ASYNC_LDS
    __builtin_amdgcn_s_wait_asynccnt(0);
#endif
    __syncthreads();
#pragma unroll 2
    for (int it = 0; it < NIT; ++it) {
        if (it + 1 < NIT) loadStage((it + 1) * BK);
        compute(it & 1);
        if (it + 1 < NIT) {
            storeStage((it + 1) & 1, (it + 1) * BK);
#if USE_ASYNC_LDS
            __builtin_amdgcn_s_wait_asynccnt(0);
#endif
            __syncthreads();
        }
    }

    // ---- epilogue: scatter back to token order, scaled by top_prob ----
#pragma unroll
    for (int mi = 0; mi < 2; ++mi) {
        const int mb = tileM * BM + wm + mi * 16 + (lane >> 4) * 8;
        int   stok[8];
        float pr[8];
#pragma unroll
        for (int r = 0; r < 8; ++r) {
            int m = mb + r;
            if (m < cnt) {
                stok[r] = toklist[group * CAP + m];
                pr[r]   = top_prob[b * S + stok[r]];
            } else {
                stok[r] = -1; pr[r] = 0.f;
            }
        }
#pragma unroll
        for (int t = 0; t < 4; ++t) {
            int n = tileN * BN + wn + t * 16 + (lane & 15);
#pragma unroll
            for (int r = 0; r < 8; ++r) {
                if (stok[r] >= 0)
                    out[((size_t)(b * S + stok[r])) * DD + n] =
                        pr[r] * acc[mi][t][r];
            }
        }
    }
}

// ============================================================================
// Host launcher
// ============================================================================
extern "C" void kernel_launch(void* const* d_in, const int* in_sizes, int n_in,
                              void* d_out, int out_size, void* d_ws, size_t ws_size,
                              hipStream_t stream) {
    const float* hidden = (const float*)d_in[0];   // [B,S,D]
    const float* gate_w = (const float*)d_in[1];   // [E,D]
    const float* w1     = (const float*)d_in[2];   // [E,D,F]
    const float* w2     = (const float*)d_in[3];   // [E,F,D]

    float* out        = (float*)d_out;                          // [B,S,D]
    float* logits_out = out + (size_t)B * S * DD;               // [B,S,E]
    int*   eidx_out   = (int*)(logits_out + (size_t)B * S * E); // [B,S] int32

    // workspace carve-up
    char* w = (char*)d_ws;
    float* top_prob  = (float*)w;  w += (size_t)B * S * sizeof(float);
    int*   expert_ix = (int*)w;    w += (size_t)B * S * sizeof(int);
    int*   toklist   = (int*)w;    w += (size_t)B * E * CAP * sizeof(int);
    int*   count     = (int*)w;    w += (size_t)B * E * sizeof(int);
    w = (char*)(((uintptr_t)w + 255) & ~(uintptr_t)255);
    __bf16* hbuf = (__bf16*)w;                                  // [B*E, CAP, F] bf16 ~80MB

    moe_router<<<(B * S) / 8, 256, 0, stream>>>(hidden, gate_w, logits_out,
                                                expert_ix, top_prob);
    moe_route_scan<<<1, 32, 0, stream>>>(expert_ix, toklist, count, eidx_out);
    moe_default_out<<<B * S, 256, 0, stream>>>(hidden, top_prob, out);
    moe_gemm1<<<dim3(FF / BN, MT, B * E), 256, 0, stream>>>(
        hidden, w1, toklist, count, hbuf);
    moe_gemm2<<<dim3(DD / BN, MT, B * E), 256, 0, stream>>>(
        hbuf, w2, toklist, count, top_prob, out);
}
